// torch_SBML_kinetic_model_31602369364718
// MI455X (gfx1250) — matrix operand order repaired and checked by reading, not compiled, not run on hardware
//
#include <hip/hip_runtime.h>

// CDNA5 / gfx1250: wave32, WMMA 16x16x32 f16 -> f32 accumulate.
typedef __attribute__((ext_vector_type(16))) _Float16 v16h;
typedef __attribute__((ext_vector_type(8)))  float    v8f;
typedef __attribute__((ext_vector_type(4)))  int      v4i;

#define B_DIM 1024
#define M_DIM 2048
#define R_DIM 8192

union Frag16 { unsigned int u[8]; v16h h; };
union H2     { _Float16 h[2]; unsigned int u; };

// gfx1250 async global->LDS copy (ASYNCcnt path). Guarded: falls back to
// plain load + ds_store if the toolchain doesn't declare the builtin.
#if defined(__has_builtin)
#  if __has_builtin(__builtin_amdgcn_global_load_async_to_lds_b128)
#    define USE_ASYNC_LDS 1
#  endif
#endif
#ifndef USE_ASYNC_LDS
#  define USE_ASYNC_LDS 0
#endif

__device__ __forceinline__ void wait_stage() {
#if USE_ASYNC_LDS
    asm volatile("s_wait_asynccnt 0x0" ::: "memory");
#endif
}

// ---------------------------------------------------------------------------
// Pre-pass 0: logc = log(conc), f16  (computed once, reused across all R)
// ---------------------------------------------------------------------------
__global__ __launch_bounds__(256)
void log_kernel(const float* __restrict__ conc, _Float16* __restrict__ logc) {
    int i = blockIdx.x * 256 + threadIdx.x;
    logc[i] = (_Float16)__logf(conc[i]);
}

// ---------------------------------------------------------------------------
// Pre-pass 1: Et[r, m] = (f16) E[m, r]   (transpose via LDS, coalesced both ways)
// ---------------------------------------------------------------------------
__global__ __launch_bounds__(256)
void transpose_convert_E(const int* __restrict__ E, _Float16* __restrict__ Et) {
    __shared__ _Float16 tile[32][33];
    const int tx = threadIdx.x & 31;
    const int ty = threadIdx.x >> 5;        // 0..7
    const int r0 = blockIdx.x * 32;
    const int m0 = blockIdx.y * 32;
#pragma unroll
    for (int i = 0; i < 32; i += 8)
        tile[ty + i][tx] = (_Float16)(float)E[(size_t)(m0 + ty + i) * R_DIM + r0 + tx];
    __syncthreads();
#pragma unroll
    for (int i = 0; i < 32; i += 8)
        Et[(size_t)(r0 + ty + i) * M_DIM + m0 + tx] = tile[tx][ty + i];
}

// ---------------------------------------------------------------------------
// Pre-pass 2: Sh = (f16) S   (flat convert, 8 elements / thread, packed stores)
// ---------------------------------------------------------------------------
__global__ __launch_bounds__(256)
void convert_S(const int* __restrict__ S, _Float16* __restrict__ Sh) {
    size_t i = ((size_t)blockIdx.x * 256 + threadIdx.x) * 8;
    int4 a = *reinterpret_cast<const int4*>(S + i);
    int4 b = *reinterpret_cast<const int4*>(S + i + 4);
    H2 p0, p1, p2, p3;
    p0.h[0] = (_Float16)(float)a.x;  p0.h[1] = (_Float16)(float)a.y;
    p1.h[0] = (_Float16)(float)a.z;  p1.h[1] = (_Float16)(float)a.w;
    p2.h[0] = (_Float16)(float)b.x;  p2.h[1] = (_Float16)(float)b.y;
    p3.h[0] = (_Float16)(float)b.z;  p3.h[1] = (_Float16)(float)b.w;
    uint4 o; o.x = p0.u; o.y = p1.u; o.z = p2.u; o.w = p3.u;
    *reinterpret_cast<uint4*>(Sh + i) = o;
}

// ---------------------------------------------------------------------------
// Tile staging: pure 16B copies global->LDS (async on gfx1250 if available).
// A tile: 128 rows x 32 K ; B tile: 128 cols x 32 K (both K-contiguous).
// 512 + 512 uint4 over 256 threads -> 2+2 copies per thread.
// ---------------------------------------------------------------------------
__device__ __forceinline__ void stage_pair(const _Float16* __restrict__ gA,
                                           const _Float16* __restrict__ gB,
                                           _Float16* sA, _Float16* sB,
                                           int t, int rowBase, int colBase,
                                           int kt, int KTOT) {
#pragma unroll
    for (int it = 0; it < 2; ++it) {
        int idx = t + it * 256;
        int row = idx >> 2;             // 0..127
        int kq  = (idx & 3) * 8;        // 0,8,16,24 halves (16B)
        const _Float16* srcA = gA + (size_t)(rowBase + row) * KTOT + kt + kq;
        const _Float16* srcB = gB + (size_t)(colBase + row) * KTOT + kt + kq;
        _Float16* dA = sA + row * 32 + kq;
        _Float16* dB = sB + row * 32 + kq;
#if USE_ASYNC_LDS
        __builtin_amdgcn_global_load_async_to_lds_b128(
            (__attribute__((address_space(1))) v4i*)srcA,
            (__attribute__((address_space(3))) v4i*)dA, 0, 0);
        __builtin_amdgcn_global_load_async_to_lds_b128(
            (__attribute__((address_space(1))) v4i*)srcB,
            (__attribute__((address_space(3))) v4i*)dB, 0, 0);
#else
        *reinterpret_cast<uint4*>(dA) = *reinterpret_cast<const uint4*>(srcA);
        *reinterpret_cast<uint4*>(dB) = *reinterpret_cast<const uint4*>(srcB);
#endif
    }
}

// ---------------------------------------------------------------------------
// Fragment loads (ISA 7.12.2 wave32 layouts); LDS A row-major [row][32],
// LDS B column-major [col][32]: every fragment dword = one aligned b32 load.
// ---------------------------------------------------------------------------
__device__ __forceinline__ void load_a_frag(const unsigned int* As32, int row,
                                            int laneHiK, Frag16& f) {
#pragma unroll
    for (int j = 0; j < 8; ++j) {
        int kk = ((j & 4) ? 16 : 0) + laneHiK + 2 * (j & 3);
        f.u[j] = As32[row * 16 + (kk >> 1)];
    }
}
__device__ __forceinline__ void load_b_frag(const unsigned int* Bs32, int col,
                                            int laneHiK, Frag16& f) {
#pragma unroll
    for (int j = 0; j < 8; ++j) {
        int kk = ((j & 4) ? 16 : 0) + laneHiK + 2 * (j & 3);
        f.u[j] = Bs32[col * 16 + (kk >> 1)];
    }
}

// ---------------------------------------------------------------------------
// Shared WMMA GEMM:  out[b, n] = sum_k A[b, k] * Bt[n, k]
//   Workgroup: 256 thr = 8 waves, tile 128x128; wave tile 32x64
//   (2 A-frags x 4 B-frags -> 8 v_wmma per K-step of 32); LDS double-buffered.
//   RATE epilogue: outh = (f16) k[n] * exp(acc)   else: outf = acc (f32)
// ---------------------------------------------------------------------------
template <bool RATE>
__global__ __launch_bounds__(256)
void gemm_ws(const _Float16* __restrict__ A, const _Float16* __restrict__ Bt,
             const float* __restrict__ kvec, _Float16* __restrict__ outh,
             float* __restrict__ outf, int KTOT, int NTOT) {
    __shared__ __align__(16) _Float16 As[2][128 * 32];
    __shared__ __align__(16) _Float16 Bs[2][128 * 32];

    const int t    = threadIdx.x;
    const int lane = t & 31;
    const int wave = t >> 5;
    const int wrow = (wave >> 1) * 32;   // 0,32,64,96
    const int wcol = (wave & 1) * 64;    // 0,64
    const int rowBase = blockIdx.y * 128;
    const int colBase = blockIdx.x * 128;

    v8f acc[2][4] = {};

    stage_pair(A, Bt, As[0], Bs[0], t, rowBase, colBase, 0, KTOT);
    wait_stage();
    __syncthreads();

    int cur = 0;
    for (int kt = 0; kt < KTOT; kt += 32) {
        if (kt + 32 < KTOT)
            stage_pair(A, Bt, As[cur ^ 1], Bs[cur ^ 1], t, rowBase, colBase,
                       kt + 32, KTOT);

        const unsigned int* As32 = reinterpret_cast<const unsigned int*>(As[cur]);
        const unsigned int* Bs32 = reinterpret_cast<const unsigned int*>(Bs[cur]);
        const int mrow = lane & 15;
        const int hiKA = (lane & 16) ? 8  : 0;
        const int hiKB = (lane & 16) ? 8  : 0;

        Frag16 a[2], b[4];
#pragma unroll
        for (int mt = 0; mt < 2; ++mt)
            load_a_frag(As32, wrow + mt * 16 + mrow, hiKA, a[mt]);
#pragma unroll
        for (int nt = 0; nt < 4; ++nt)
            load_b_frag(Bs32, wcol + nt * 16 + mrow, hiKB, b[nt]);

#pragma unroll
        for (int mt = 0; mt < 2; ++mt)
#pragma unroll
            for (int nt = 0; nt < 4; ++nt)
                acc[mt][nt] = __builtin_amdgcn_wmma_f32_16x16x32_f16(
                    false, a[mt].h, false, b[nt].h, (short)0, acc[mt][nt],
                    false, false);

        wait_stage();
        __syncthreads();
        cur ^= 1;
    }

    // epilogue (C layout: VGPR i -> M = i + (laneHi?8:0), N = lane&15)
    const int n    = lane & 15;
    const int mofs = (lane & 16) ? 8 : 0;
#pragma unroll
    for (int mt = 0; mt < 2; ++mt) {
#pragma unroll
        for (int nt = 0; nt < 4; ++nt) {
            int gcol = colBase + wcol + nt * 16 + n;
            if constexpr (RATE) {
                float kv = kvec[gcol];
#pragma unroll
                for (int i = 0; i < 8; ++i) {
                    int grow = rowBase + wrow + mt * 16 + mofs + i;
                    outh[(size_t)grow * NTOT + gcol] =
                        (_Float16)(kv * __expf(acc[mt][nt][i]));
                }
            } else {
#pragma unroll
                for (int i = 0; i < 8; ++i) {
                    int grow = rowBase + wrow + mt * 16 + mofs + i;
                    outf[(size_t)grow * NTOT + gcol] = acc[mt][nt][i];
                }
            }
        }
    }
}

// ---------------------------------------------------------------------------
extern "C" void kernel_launch(void* const* d_in, const int* in_sizes, int n_in,
                              void* d_out, int out_size, void* d_ws, size_t ws_size,
                              hipStream_t stream) {
    const float* conc = (const float*)d_in[0];   // [B, M] f32
    const int*   E    = (const int*)d_in[1];     // [M, R] i32
    const int*   S    = (const int*)d_in[2];     // [M, R] i32
    const float* kvec = (const float*)d_in[3];   // [R]    f32
    float*       out  = (float*)d_out;           // [B, M] f32

    // ws layout (bytes): logc 4MB | v 16MB | Et 32MB | Sh 32MB  = 84MB
    char* w = (char*)d_ws;
    _Float16* logc = (_Float16*)(w);
    _Float16* vbuf = (_Float16*)(w + (size_t)B_DIM * M_DIM * 2);
    _Float16* Et   = (_Float16*)(w + (size_t)B_DIM * M_DIM * 2 + (size_t)B_DIM * R_DIM * 2);
    _Float16* Sh   = (_Float16*)(w + (size_t)B_DIM * M_DIM * 2 + (size_t)B_DIM * R_DIM * 2
                                   + (size_t)M_DIM * R_DIM * 2);

    log_kernel<<<(B_DIM * M_DIM) / 256, 256, 0, stream>>>(conc, logc);
    transpose_convert_E<<<dim3(R_DIM / 32, M_DIM / 32), 256, 0, stream>>>(E, Et);
    convert_S<<<((size_t)M_DIM * R_DIM) / 2048, 256, 0, stream>>>(S, Sh);

    // GEMM1: v = k * exp(logc @ E)   [1024 x 8192], K = 2048
    gemm_ws<true><<<dim3(R_DIM / 128, B_DIM / 128), 256, 0, stream>>>(
        logc, Et, kvec, vbuf, nullptr, M_DIM, R_DIM);
    // GEMM2: dXdt = v @ S^T          [1024 x 2048], K = 8192
    gemm_ws<false><<<dim3(M_DIM / 128, B_DIM / 128), 256, 0, stream>>>(
        vbuf, Sh, nullptr, nullptr, out, R_DIM, M_DIM);
}
